// CodePredBlock_8650064134908
// MI455X (gfx1250) — matrix operand order repaired and checked by reading, not compiled, not run on hardware
//
#include <hip/hip_runtime.h>
#include <hip/hip_bf16.h>
#include <math.h>

// ---------------------------------------------------------------------------
// MI455X (gfx1250) transformer block: bf16 WMMA GEMMs + flash attention.
// All matmuls use v_wmma_f32_16x16x32_bf16 (fp32 accumulate).
// ---------------------------------------------------------------------------

typedef __attribute__((ext_vector_type(16))) __bf16 v16bf;
typedef __attribute__((ext_vector_type(8)))  __bf16 bf16x8;
typedef __attribute__((ext_vector_type(8)))  float  v8f;

#define EPSV 1e-6f

constexpr int Bc = 2, Tc = 2048, Hc = 2048, NHc = 16, NKVc = 8, HDc = 128, DFFc = 8192;
constexpr int MTc = Bc * Tc;   // 4096 rows

__device__ __forceinline__ v16bf make_frag(bf16x8 lo, bf16x8 hi8) {
  v16bf r;
#pragma unroll
  for (int i = 0; i < 8; ++i) { r[i] = lo[i]; r[i + 8] = hi8[i]; }
  return r;
}

__device__ __forceinline__ v8f wmma_bf16(v16bf a, v16bf b, v8f c) {
  return __builtin_amdgcn_wmma_f32_16x16x32_bf16(false, a, false, b, (short)0, c, false, false);
}

// ---------------------------------------------------------------------------
// fp32 -> bf16 conversion (weights)
// ---------------------------------------------------------------------------
__global__ void cvt_bf16_kernel(const float* __restrict__ in, __bf16* __restrict__ out, long n) {
  long i = (long)blockIdx.x * blockDim.x + threadIdx.x;
  long stride = (long)gridDim.x * blockDim.x;
  for (; i < n; i += stride) out[i] = (__bf16)in[i];
}

// ---------------------------------------------------------------------------
// RMSNorm over width, write bf16 (one block per row)
// ---------------------------------------------------------------------------
__global__ __launch_bounds__(256) void rmsnorm_bf16_kernel(
    const float* __restrict__ x, const float* __restrict__ w,
    __bf16* __restrict__ out, int width)
{
  const int row = blockIdx.x;
  const size_t base = (size_t)row * width;
  const int tid = threadIdx.x;
  float ss = 0.f;
  for (int i = tid; i < width; i += 256) { float v = x[base + i]; ss += v * v; }
#pragma unroll
  for (int m = 16; m >= 1; m >>= 1) ss += __shfl_xor(ss, m, 32);
  __shared__ float red[8];
  __shared__ float snorm;
  if ((tid & 31) == 0) red[tid >> 5] = ss;
  __syncthreads();
  if (tid == 0) {
    float t = 0.f;
    for (int j = 0; j < 8; ++j) t += red[j];
    snorm = rsqrtf(t / (float)width + EPSV);
  }
  __syncthreads();
  const float rn = snorm;
  for (int i = tid; i < width; i += 256)
    out[base + i] = (__bf16)(x[base + i] * rn * w[i]);
}

// ---------------------------------------------------------------------------
// GEMM: C[M,N] (f32) = A[M,K](bf16,row-major) * W[N,K](bf16,row-major)^T (+res)
// Block 256 thr = 8 waves; tile 128x128x64; per-wave 64x32 (8 wmma accs).
// LDS stride 72 bf16 (padded) -> conflict-free ds_load_b128 fragment reads.
// Register-staged global prefetch; launch_bounds(.,1) keeps it out of scratch.
// ---------------------------------------------------------------------------
#define GBM 128
#define GBN 128
#define GBK 64
#define GSTR 72

__global__ __launch_bounds__(256, 1) void gemm_bf16_kernel(
    const __bf16* __restrict__ A, const __bf16* __restrict__ W,
    float* __restrict__ C, const float* __restrict__ res,
    int M, int N, int K)
{
  __shared__ __bf16 sA[GBM * GSTR];
  __shared__ __bf16 sW[GBN * GSTR];

  const int tid = threadIdx.x;
  const int lane = tid & 31;
  const int wid  = tid >> 5;
  const int lm = lane & 15;
  const int hl = lane >> 4;
  const int wm = wid >> 2;          // 0..1  (64-row slice)
  const int wn = wid & 3;           // 0..3  (32-col slice)
  const int m0 = blockIdx.y * GBM;
  const int n0 = blockIdx.x * GBN;

  v8f acc[4][2];
#pragma unroll
  for (int a = 0; a < 4; ++a)
#pragma unroll
    for (int b = 0; b < 2; ++b)
#pragma unroll
      for (int e = 0; e < 8; ++e) acc[a][b][e] = 0.f;

  // ---- staging geometry: each thread owns rows {r0, r0+32, r0+64, r0+96},
  // one 16B chunk (8 bf16) at column cc*8 within the 64-wide K tile.
  const int r0 = tid >> 3;          // 0..31
  const int cc = tid & 7;           // 0..7
  const __bf16* gA = A + (size_t)(m0 + r0) * K + cc * 8;
  const __bf16* gW = W + (size_t)(n0 + r0) * K + cc * 8;
  const size_t rowK32 = (size_t)32 * K;
  __bf16* lA = &sA[r0 * GSTR + cc * 8];
  __bf16* lW = &sW[r0 * GSTR + cc * 8];

  uint4 ra0, ra1, ra2, ra3, rw0, rw1, rw2, rw3;

#define GEMM_GLOAD(kt_)                                                        \
  do {                                                                         \
    const __bf16* pA_ = gA + (size_t)(kt_) * GBK;                              \
    const __bf16* pW_ = gW + (size_t)(kt_) * GBK;                              \
    ra0 = *(const uint4*)(pA_);                                                \
    ra1 = *(const uint4*)(pA_ + rowK32);                                       \
    ra2 = *(const uint4*)(pA_ + 2 * rowK32);                                   \
    ra3 = *(const uint4*)(pA_ + 3 * rowK32);                                   \
    rw0 = *(const uint4*)(pW_);                                                \
    rw1 = *(const uint4*)(pW_ + rowK32);                                       \
    rw2 = *(const uint4*)(pW_ + 2 * rowK32);                                   \
    rw3 = *(const uint4*)(pW_ + 3 * rowK32);                                   \
  } while (0)

#define GEMM_LSTORE()                                                          \
  do {                                                                         \
    *(uint4*)(lA)             = ra0;                                           \
    *(uint4*)(lA + 32 * GSTR) = ra1;                                           \
    *(uint4*)(lA + 64 * GSTR) = ra2;                                           \
    *(uint4*)(lA + 96 * GSTR) = ra3;                                           \
    *(uint4*)(lW)             = rw0;                                           \
    *(uint4*)(lW + 32 * GSTR) = rw1;                                           \
    *(uint4*)(lW + 64 * GSTR) = rw2;                                           \
    *(uint4*)(lW + 96 * GSTR) = rw3;                                           \
  } while (0)

  const int nk = K / GBK;
  GEMM_GLOAD(0);
  GEMM_LSTORE();

  // fragment base pointers (immediate ds offsets inside the unrolled loops)
  const __bf16* paBase = &sA[(wm * 64 + lm) * GSTR];
  const __bf16* pbBase = &sW[(wn * 32 + lm) * GSTR];
  const int aoff = hl * 8;    // A frag: chunks at +aoff and +aoff+16
  const int boff = hl * 8;    // B frag: chunks at +boff and +boff+16

  for (int kt = 0; kt < nk; ++kt) {
    __syncthreads();                         // LDS writes visible
    if (kt + 1 < nk) GEMM_GLOAD(kt + 1);     // prefetch next tile into regs
#pragma unroll
    for (int ks = 0; ks < 2; ++ks) {         // two K=32 wmma steps per tile
      v16bf af[4], wf[2];
#pragma unroll
      for (int mt = 0; mt < 4; ++mt) {
        const __bf16* pa = paBase + mt * 16 * GSTR + ks * 32 + aoff;
        af[mt] = make_frag(*(const bf16x8*)pa, *(const bf16x8*)(pa + 16));
      }
#pragma unroll
      for (int nt = 0; nt < 2; ++nt) {
        const __bf16* pb = pbBase + nt * 16 * GSTR + ks * 32 + boff;
        wf[nt] = make_frag(*(const bf16x8*)pb, *(const bf16x8*)(pb + 16));
      }
#pragma unroll
      for (int mt = 0; mt < 4; ++mt)
#pragma unroll
        for (int nt = 0; nt < 2; ++nt)
          acc[mt][nt] = wmma_bf16(af[mt], wf[nt], acc[mt][nt]);
    }
    __syncthreads();                         // all LDS reads done
    if (kt + 1 < nk) GEMM_LSTORE();
  }

  // Epilogue: C layout (m = r + 8*hl, n = lm) per 16x16 tile
#pragma unroll
  for (int mt = 0; mt < 4; ++mt) {
#pragma unroll
    for (int nt = 0; nt < 2; ++nt) {
      const int gn = n0 + wn * 32 + nt * 16 + lm;
#pragma unroll
      for (int r = 0; r < 8; ++r) {
        const int gm = m0 + wm * 64 + mt * 16 + r + 8 * hl;
        const size_t idx = (size_t)gm * N + gn;
        float v = acc[mt][nt][r];
        if (res) v += res[idx];
        C[idx] = v;
      }
    }
  }
#undef GEMM_GLOAD
#undef GEMM_LSTORE
}

// ---------------------------------------------------------------------------
// Per-head RMSNorm + NeoX RoPE, fp32 in -> bf16 [B, nheads, T, HD] out
// grid (B*T, nheads), 128 threads (= HD)
// ---------------------------------------------------------------------------
__global__ __launch_bounds__(128) void qk_rope_kernel(
    const float* __restrict__ src, const float* __restrict__ nw,
    const float* __restrict__ cosp, const float* __restrict__ sinp,
    __bf16* __restrict__ dst, int nheads)
{
  const int row = blockIdx.x;              // b*T + t
  const int h = blockIdx.y;
  const int d = threadIdx.x;               // 0..127
  const int t = row & (Tc - 1);
  const int b = row >> 11;
  float v = src[(size_t)row * nheads * HDc + h * HDc + d];
  float ss = v * v;
#pragma unroll
  for (int m = 16; m >= 1; m >>= 1) ss += __shfl_xor(ss, m, 32);
  __shared__ float red[4];
  __shared__ float snorm;
  __shared__ float sh[HDc];
  if ((d & 31) == 0) red[d >> 5] = ss;
  __syncthreads();
  if (d == 0) snorm = rsqrtf((red[0] + red[1] + red[2] + red[3]) / (float)HDc + EPSV);
  __syncthreads();
  sh[d] = v * snorm * nw[d];
  __syncthreads();
  float o;
  if (d < 64) {
    float c = cosp[t * 64 + d], s = sinp[t * 64 + d];
    o = sh[d] * c - sh[d + 64] * s;
  } else {
    int dd = d - 64;
    float c = cosp[t * 64 + dd], s = sinp[t * 64 + dd];
    o = sh[dd] * s + sh[d] * c;
  }
  dst[(((size_t)(b * nheads + h)) * Tc + t) * HDc + d] = (__bf16)o;
}

// V: fp32 [B*T, NKV*HD] -> bf16 [B, NKV, T, HD]
__global__ void v_pack_kernel(const float* __restrict__ src, __bf16* __restrict__ dst) {
  const long n = (long)MTc * NKVc * HDc;
  long i = (long)blockIdx.x * blockDim.x + threadIdx.x;
  const long stride = (long)gridDim.x * blockDim.x;
  for (; i < n; i += stride) {
    int d = (int)(i & 127);
    int h = (int)((i >> 7) & (NKVc - 1));
    long row = i >> 10;
    int t = (int)(row & (Tc - 1));
    int b = (int)(row >> 11);
    dst[(((size_t)(b * NKVc + h)) * Tc + t) * HDc + d] = (__bf16)src[i];
  }
}

// ---------------------------------------------------------------------------
// Flash attention (causal, GQA rep=2): Q[B,NH,T,HD], K/V[B,NKV,T,HD] bf16
// 128 thr = 4 waves; block = 64 q rows (16 per wave); KV tiles of 32.
// Scores/probs never touch HBM. ctx out bf16 [B,T,NH*HD].
// ---------------------------------------------------------------------------
#define FA_KSTR 136
#define FA_VSTR 40
#define FA_PSTR 40

__global__ __launch_bounds__(128, 1) void flash_attn_kernel(
    const __bf16* __restrict__ Q, const __bf16* __restrict__ Kd,
    const __bf16* __restrict__ Vd, __bf16* __restrict__ ctx)
{
  __shared__ __bf16 sK[32 * FA_KSTR];       // K tile, row-major
  __shared__ __bf16 sV[128 * FA_VSTR];      // V tile, transposed: [d][kv]
  __shared__ __bf16 sP[4][16 * FA_PSTR];    // per-wave P relayout buffer

  const int tid = threadIdx.x;
  const int lane = tid & 31;
  const int wid = tid >> 5;
  const int lm = lane & 15;
  const int hl = lane >> 4;

  const int bh = blockIdx.x;
  const int b = bh / NHc;
  const int h = bh % NHc;
  const int kvh = h >> 1;                    // NH/NKV = 2
  const int qtile = blockIdx.y;
  const int qrow0 = qtile * 64 + wid * 16;
  const float scale = 0.08838834764831845f;  // 1/sqrt(128)

  // Q fragments (4 x K=32 steps over HD=128), kept in registers
  const __bf16* qptr = Q + (((size_t)bh) * Tc + qrow0 + lm) * HDc;
  v16bf qf[4];
#pragma unroll
  for (int kc = 0; kc < 4; ++kc) {
    const __bf16* p = qptr + kc * 32 + hl * 8;
    qf[kc] = make_frag(*(const bf16x8*)p, *(const bf16x8*)(p + 16));
  }

  float mrun[8], lrun[8];
  v8f o[8];
#pragma unroll
  for (int r = 0; r < 8; ++r) { mrun[r] = -3.0e38f; lrun[r] = 0.f; }
#pragma unroll
  for (int dt = 0; dt < 8; ++dt)
#pragma unroll
    for (int e = 0; e < 8; ++e) o[dt][e] = 0.f;

  const size_t kvb = ((size_t)(b * NKVc + kvh)) * Tc;
  const int ntiles = qtile * 2 + 2;          // cover kv <= qrow0+63

  // hoisted staging pointers
  const int krk = tid >> 4;                  // 0..7 base row
  const int kcc = tid & 15;                  // 16B chunk within 128-wide row
  const int vr = tid & 31;
  const int dg = tid >> 5;
  __bf16* sKp = &sK[krk * FA_KSTR + kcc * 8];
  const __bf16* pk0Base = &sK[lm * FA_KSTR + hl * 8];
  const __bf16* pk1Base = &sK[(16 + lm) * FA_KSTR + hl * 8];
  const __bf16* pvBase = &sV[lm * FA_VSTR + hl * 8];
  const __bf16* pp = &sP[wid][lm * FA_PSTR + hl * 8];

  for (int it = 0; it < ntiles; ++it) {
    const int kv0 = it * 32;
    __syncthreads();                         // previous tile's LDS reads done
    // stage K tile (32x128): 512 x 16B chunks / 128 threads (rows krk+{0,8,16,24})
    {
      const __bf16* gk = Kd + (kvb + kv0 + krk) * HDc + kcc * 8;
#pragma unroll
      for (int c = 0; c < 4; ++c)
        *(uint4*)(sKp + c * 8 * FA_KSTR) = *(const uint4*)(gk + (size_t)c * 8 * HDc);
    }
    // stage V tile transposed
    {
      const __bf16* vp = Vd + (kvb + kv0 + vr) * HDc + dg * 32;
#pragma unroll
      for (int j = 0; j < 32; ++j)
        sV[(dg * 32 + j) * FA_VSTR + vr] = vp[j];
    }
    __syncthreads();

    // S = Q K^T  (two 16-col n-tiles)
    v8f s0 = {0.f,0.f,0.f,0.f,0.f,0.f,0.f,0.f};
    v8f s1 = {0.f,0.f,0.f,0.f,0.f,0.f,0.f,0.f};
#pragma unroll
    for (int kc = 0; kc < 4; ++kc) {
      const __bf16* pk0 = pk0Base + kc * 32;
      s0 = wmma_bf16(qf[kc], make_frag(*(const bf16x8*)pk0, *(const bf16x8*)(pk0 + 16)), s0);
      const __bf16* pk1 = pk1Base + kc * 32;
      s1 = wmma_bf16(qf[kc], make_frag(*(const bf16x8*)pk1, *(const bf16x8*)(pk1 + 16)), s1);
    }

    // online softmax (row m = r + 8*hl lives across 16 lanes -> shfl reduce)
    float alpha[8];
#pragma unroll
    for (int r = 0; r < 8; ++r) {
      const int qr = qrow0 + r + 8 * hl;
      float a0 = s0[r] * scale;
      float a1 = s1[r] * scale;
      if (kv0 + lm > qr)      a0 = -1.0e30f;
      if (kv0 + 16 + lm > qr) a1 = -1.0e30f;
      float mx = fmaxf(a0, a1);
#pragma unroll
      for (int mk = 8; mk >= 1; mk >>= 1) mx = fmaxf(mx, __shfl_xor(mx, mk, 32));
      const float nm = fmaxf(mrun[r], mx);
      const float p0 = __expf(a0 - nm);
      const float p1 = __expf(a1 - nm);
      float rs = p0 + p1;
#pragma unroll
      for (int mk = 8; mk >= 1; mk >>= 1) rs += __shfl_xor(rs, mk, 32);
      alpha[r] = __expf(mrun[r] - nm);
      lrun[r] = lrun[r] * alpha[r] + rs;
      mrun[r] = nm;
      s0[r] = p0; s1[r] = p1;
    }
#pragma unroll
    for (int dt = 0; dt < 8; ++dt)
#pragma unroll
      for (int r = 0; r < 8; ++r) o[dt][r] *= alpha[r];

    // relayout P (C-layout -> A-fragment layout) through per-wave LDS.
    // Same-wave DS ops are in-order (ISA 7.3) -> no barrier needed.
#pragma unroll
    for (int r = 0; r < 8; ++r) {
      const int pr = r + 8 * hl;
      sP[wid][pr * FA_PSTR + lm]      = (__bf16)s0[r];
      sP[wid][pr * FA_PSTR + 16 + lm] = (__bf16)s1[r];
    }
    v16bf pf = make_frag(*(const bf16x8*)pp, *(const bf16x8*)(pp + 16));

    // O += P V   (K=32: exactly one wmma per 16-wide hd tile)
#pragma unroll
    for (int dt = 0; dt < 8; ++dt) {
      const __bf16* vb = pvBase + dt * 16 * FA_VSTR;
      o[dt] = wmma_bf16(pf, make_frag(*(const bf16x8*)vb, *(const bf16x8*)(vb + 16)), o[dt]);
    }
  }

  float invl[8];
#pragma unroll
  for (int r = 0; r < 8; ++r) invl[r] = 1.f / lrun[r];
#pragma unroll
  for (int dt = 0; dt < 8; ++dt) {
#pragma unroll
    for (int r = 0; r < 8; ++r) {
      const int t = qrow0 + r + 8 * hl;
      ctx[((size_t)(b * Tc + t)) * (NHc * HDc) + h * HDc + dt * 16 + lm] =
          (__bf16)(o[dt][r] * invl[r]);
    }
  }
}

// ---------------------------------------------------------------------------
// h = silu(gate) * up  -> bf16
// ---------------------------------------------------------------------------
__global__ void swiglu_kernel(const float* __restrict__ g, const float* __restrict__ u,
                              __bf16* __restrict__ outb, long n) {
  long i = (long)blockIdx.x * blockDim.x + threadIdx.x;
  const long stride = (long)gridDim.x * blockDim.x;
  for (; i < n; i += stride) {
    float gv = g[i];
    float sg = gv / (1.f + __expf(-gv));
    outb[i] = (__bf16)(sg * u[i]);
  }
}

// ---------------------------------------------------------------------------
extern "C" void kernel_launch(void* const* d_in, const int* in_sizes, int n_in,
                              void* d_out, int out_size, void* d_ws, size_t ws_size,
                              hipStream_t stream)
{
  const float* x   = (const float*)d_in[0];
  // d_in[1] = attn_mask (unused; causal mask applied analytically)
  const float* anw = (const float*)d_in[2];
  const float* fnw = (const float*)d_in[3];
  const float* wq  = (const float*)d_in[4];
  const float* wk  = (const float*)d_in[5];
  const float* wv  = (const float*)d_in[6];
  const float* wo  = (const float*)d_in[7];
  const float* qnw = (const float*)d_in[8];
  const float* knw = (const float*)d_in[9];
  const float* wg  = (const float*)d_in[10];
  const float* wu  = (const float*)d_in[11];
  const float* wd  = (const float*)d_in[12];
  const float* rc  = (const float*)d_in[13];
  const float* rs  = (const float*)d_in[14];
  float* out = (float*)d_out;

  char* cur = (char*)d_ws;
  auto alloc = [&](size_t bytes) -> char* {
    char* r = cur;
    cur += (bytes + 255) & ~(size_t)255;
    return r;
  };
  const size_t SQ = (size_t)NHc * HDc * Hc;     // 4M elems
  const size_t SKV = (size_t)NKVc * HDc * Hc;   // 2M
  const size_t SG = (size_t)DFFc * Hc;          // 16M

  __bf16* wq_b  = (__bf16*)alloc(SQ * 2);
  __bf16* wk_b  = (__bf16*)alloc(SKV * 2);
  __bf16* wv_b  = (__bf16*)alloc(SKV * 2);
  __bf16* wo_b  = (__bf16*)alloc(SQ * 2);
  __bf16* wg_b  = (__bf16*)alloc(SG * 2);
  __bf16* wu_b  = (__bf16*)alloc(SG * 2);
  __bf16* wd_b  = (__bf16*)alloc(SG * 2);
  __bf16* xn_b  = (__bf16*)alloc((size_t)MTc * Hc * 2);
  __bf16* q_b   = (__bf16*)alloc((size_t)MTc * NHc * HDc * 2);
  __bf16* k_b   = (__bf16*)alloc((size_t)MTc * NKVc * HDc * 2);
  __bf16* v_b   = (__bf16*)alloc((size_t)MTc * NKVc * HDc * 2);
  __bf16* ctx_b = (__bf16*)alloc((size_t)MTc * NHc * HDc * 2);
  float*  x1    = (float*)alloc((size_t)MTc * Hc * 4);
  float*  bufA  = (float*)alloc((size_t)MTc * DFFc * 4);   // q/k/v f32, later gate
  float*  bufB  = (float*)alloc((size_t)MTc * DFFc * 4);   // up
  __bf16* h_b   = (__bf16*)alloc((size_t)MTc * DFFc * 2);

  float* q_f = bufA;
  float* k_f = bufA + (size_t)MTc * NHc * HDc;
  float* v_f = k_f + (size_t)MTc * NKVc * HDc;

  const int CB = 256;
  auto cvt = [&](const float* src, __bf16* dst, size_t n) {
    int blocks = (int)((n + (size_t)CB * 8 - 1) / ((size_t)CB * 8));
    cvt_bf16_kernel<<<blocks, CB, 0, stream>>>(src, dst, (long)n);
  };
  cvt(wq, wq_b, SQ);  cvt(wk, wk_b, SKV); cvt(wv, wv_b, SKV); cvt(wo, wo_b, SQ);
  cvt(wg, wg_b, SG);  cvt(wu, wu_b, SG);  cvt(wd, wd_b, SG);

  // attn pre-norm
  rmsnorm_bf16_kernel<<<MTc, 256, 0, stream>>>(x, anw, xn_b, Hc);

  // QKV projections
  gemm_bf16_kernel<<<dim3(NHc * HDc / GBN, MTc / GBM), 256, 0, stream>>>(
      xn_b, wq_b, q_f, nullptr, MTc, NHc * HDc, Hc);
  gemm_bf16_kernel<<<dim3(NKVc * HDc / GBN, MTc / GBM), 256, 0, stream>>>(
      xn_b, wk_b, k_f, nullptr, MTc, NKVc * HDc, Hc);
  gemm_bf16_kernel<<<dim3(NKVc * HDc / GBN, MTc / GBM), 256, 0, stream>>>(
      xn_b, wv_b, v_f, nullptr, MTc, NKVc * HDc, Hc);

  // q/k norm + rope -> bf16 head-major; v -> bf16 head-major
  qk_rope_kernel<<<dim3(MTc, NHc), 128, 0, stream>>>(q_f, qnw, rc, rs, q_b, NHc);
  qk_rope_kernel<<<dim3(MTc, NKVc), 128, 0, stream>>>(k_f, knw, rc, rs, k_b, NKVc);
  {
    long n = (long)MTc * NKVc * HDc;
    int blocks = (int)((n + CB * 8 - 1) / (CB * 8));
    v_pack_kernel<<<blocks, CB, 0, stream>>>(v_f, v_b);
  }

  // flash attention
  flash_attn_kernel<<<dim3(Bc * NHc, Tc / 64), 128, 0, stream>>>(q_b, k_b, v_b, ctx_b);

  // output projection + residual(x)
  gemm_bf16_kernel<<<dim3(Hc / GBN, MTc / GBM), 256, 0, stream>>>(
      ctx_b, wo_b, x1, x, MTc, Hc, NHc * HDc);

  // ffn pre-norm
  rmsnorm_bf16_kernel<<<MTc, 256, 0, stream>>>(x1, fnw, xn_b, Hc);

  // gate / up
  gemm_bf16_kernel<<<dim3(DFFc / GBN, MTc / GBM), 256, 0, stream>>>(
      xn_b, wg_b, bufA, nullptr, MTc, DFFc, Hc);
  gemm_bf16_kernel<<<dim3(DFFc / GBN, MTc / GBM), 256, 0, stream>>>(
      xn_b, wu_b, bufB, nullptr, MTc, DFFc, Hc);

  // swiglu
  {
    long n = (long)MTc * DFFc;
    int blocks = (int)((n + CB * 8 - 1) / (CB * 8));
    swiglu_kernel<<<blocks, CB, 0, stream>>>(bufA, bufB, h_b, n);
  }

  // down projection + residual(x1) -> output
  gemm_bf16_kernel<<<dim3(Hc / GBN, MTc / GBM), 256, 0, stream>>>(
      h_b, wd_b, out, x1, MTc, Hc, DFFc);
}
